// peepLSTM_76381698392571
// MI455X (gfx1250) — matrix operand order repaired and checked by reading, not compiled, not run on hardware
//
#include <hip/hip_runtime.h>
#include <math.h>

// Problem constants (from reference)
#define B_   128
#define T_   512
#define I_   128
#define H_   512
#define NC_  100
#define NG   8     // batch groups (128/16)
#define NS   16    // h-slices per group
#define HS   32    // h columns owned per WG
#define KX   128   // K of xe @ Wx
#define KC   512   // K of C  @ Wc
#define KTOT 640   // KX + KC
#define NLOC 128   // gate columns per WG (4 gates x HS)
// Padded LDS row stride: 648 bf16 = 1296 B = 324 words; 324 % 64 = 4 -> the 16
// rows read by the 16 lanes of a WMMA operand land on distinct LDS banks.
#define LSTR 648

typedef __attribute__((ext_vector_type(16))) __bf16 v16bf;
typedef __attribute__((ext_vector_type(8)))  float  v8f;
typedef __attribute__((ext_vector_type(4)))  unsigned int u32x4;
struct u32x8 { u32x4 lo, hi; };

// Assemble a 16-element bf16 WMMA operand register from two 16B chunks.
__device__ __forceinline__ v16bf ld_pair(const __bf16* p_lo, const __bf16* p_hi) {
  u32x8 u;
  u.lo = *reinterpret_cast<const u32x4*>(p_lo);
  u.hi = *reinterpret_cast<const u32x4*>(p_hi);
  return __builtin_bit_cast(v16bf, u);
}

// CDNA5 async global->LDS copy, 16B per lane (ASYNCcnt-tracked, no VGPRs).
__device__ __forceinline__ void async_cp16(const void* gsrc, void* ldst) {
  unsigned lofs = (unsigned)(unsigned long long)ldst;  // low 32b = LDS offset
  asm volatile("global_load_async_to_lds_b128 %0, %1, off"
               :: "v"(lofs), "v"(gsrc) : "memory");
}
__device__ __forceinline__ void wait_async0() {
  asm volatile("s_wait_asynccnt 0x0" ::: "memory");
}

#define LOG2E_ 1.4426950408889634f
__device__ __forceinline__ float sigm(float v) {   // v_exp_f32 + v_rcp_f32
  return __builtin_amdgcn_rcpf(1.0f + __builtin_amdgcn_exp2f(-LOG2E_ * v));
}
__device__ __forceinline__ float fast_tanh(float v) {
#if __has_builtin(__builtin_amdgcn_tanhf)
  return __builtin_amdgcn_tanhf(v);                // CDNA5 v_tanh_f32
#else
  return 1.0f - 2.0f * __builtin_amdgcn_rcpf(1.0f +
             __builtin_amdgcn_exp2f(2.0f * LOG2E_ * v));
#endif
}

// Software-pipelined K-loop: load operands for step ks+1 before issuing the
// WMMA for step ks, so ds_load latency hides behind matrix-core execution.
template <int NKS>
__device__ __forceinline__ v8f gemm_ks(const __bf16* Ap, const __bf16* Bp, v8f acc) {
  v16bf a = ld_pair(Ap, Ap + 16);
  v16bf b = ld_pair(Bp, Bp + 16);
#pragma unroll
  for (int ks = 0; ks < NKS; ++ks) {
    v16bf an = a, bn = b;
    if (ks + 1 < NKS) {
      an = ld_pair(Ap + (ks + 1) * 32, Ap + (ks + 1) * 32 + 16);
      bn = ld_pair(Bp + (ks + 1) * 32, Bp + (ks + 1) * 32 + 16);
    }
    acc = __builtin_amdgcn_wmma_f32_16x16x32_bf16(false, a, false, b,
                                                  (short)0, acc, false, false);
    a = an; b = bn;
  }
  return acc;
}

// ---------------------------------------------------------------------------
// Kernel 1: gather embeddings -> bf16 xe_all[b][t][i]; zero C double-buffer
// and group barrier state.
// ---------------------------------------------------------------------------
__global__ __launch_bounds__(256) void k_prep(const int* __restrict__ xidx,
                                              const float* __restrict__ emb,
                                              __bf16* __restrict__ xe_all,
                                              __bf16* __restrict__ cbuf,
                                              int* __restrict__ bar) {
  size_t i = (size_t)blockIdx.x * 256 + threadIdx.x;
  const size_t total = (size_t)B_ * T_ * I_;
  if (i < total) {
    size_t bt = i >> 7;            // / I_
    int    k  = (int)(i & (I_ - 1));
    int    id = xidx[bt];          // emb[0] == 0 handles padding rows
    xe_all[i] = (__bf16)emb[(size_t)id * I_ + k];
  }
  if (i < (size_t)2 * NG * 16 * KC) cbuf[i] = (__bf16)0.0f;  // both C slots = 0
  if (i < 64) bar[i] = 0;
}

// ---------------------------------------------------------------------------
// Kernel 2: persistent recurrence. 128 WGs = 8 batch groups x 16 h-slices.
// Weights pinned in LDS (bf16, B-operand layout, bank-conflict-padded).
// Per step: async-stage the 16-row A panel (xe slice + group C) into LDS,
// all 8 waves consume it with ds_load_b128 -> v_wmma_f32_16x16x32_bf16.
// ---------------------------------------------------------------------------
__global__ __launch_bounds__(256) void k_rnn(
    const int*   __restrict__ xidx,
    const float* __restrict__ Wfx, const float* __restrict__ Wfc, const float* __restrict__ bf_,
    const float* __restrict__ Wix, const float* __restrict__ Wic, const float* __restrict__ bi_,
    const float* __restrict__ Wox, const float* __restrict__ Woc, const float* __restrict__ bo_,
    const float* __restrict__ Wcx, const float* __restrict__ bc_,
    const __bf16* __restrict__ xe_all,
    __bf16* __restrict__ cbuf,        // [2][NG][16][KC] bf16
    float*  __restrict__ hf,          // [B_][H_] f32
    int*    __restrict__ bar)         // [NG][2] (count, generation)
{
  extern __shared__ char smem[];
  __bf16* Wl    = (__bf16*)smem;                                   // [NLOC][LSTR]
  __bf16* Abuf  = (__bf16*)(smem + (size_t)NLOC * LSTR * 2);       // [16][LSTR]
  float*  gbuf  = (float*)((char*)Abuf + (size_t)16 * LSTR * 2);   // [8][16][16]
  float*  Cloc  = gbuf + 8 * 16 * 16;                              // [16][HS]
  float*  biasl = Cloc + 16 * HS;                                  // [NLOC]

  const int tid  = threadIdx.x;
  const int lane = tid & 31;
  const int w    = tid >> 5;            // wave id = output n-tile (0..7)
  const int g    = blockIdx.x >> 4;     // batch group
  const int hs   = (blockIdx.x & 15) * HS;

  // ---- Pin weight slice in LDS (transpose fp32 [k][h] -> bf16 [n][k]) ----
  for (int e = tid; e < NLOC * KTOT; e += 256) {
    int n = e / KTOT, k = e - n * KTOT;
    int q = n >> 5;                     // 0=f 1=i 2=o 3=c~
    int h = hs + (n & 31);
    const float* Wx = (q == 0) ? Wfx : (q == 1) ? Wix : (q == 2) ? Wox : Wcx;
    float wv = 0.0f;
    if (k < KX)      wv = Wx[(size_t)k * H_ + h];
    else if (q < 3)  wv = ((q == 0) ? Wfc : (q == 1) ? Wic : Woc)[(size_t)(k - KX) * H_ + h];
    Wl[(size_t)n * LSTR + k] = (__bf16)wv;
  }
  for (int e = tid; e < NLOC; e += 256) {
    int q = e >> 5;
    biasl[e] = ((q == 0) ? bf_ : (q == 1) ? bi_ : (q == 2) ? bo_ : bc_)[hs + (e & 31)];
  }
  for (int e = tid; e < 16 * HS; e += 256) Cloc[e] = 0.0f;
  __syncthreads();

  // ---- Per-lane WMMA operand addressing (ISA 7.12.2 layouts) ----
  const int m      = lane & 15;         // A row / D column index
  const int laneHi = lane >> 4;
  const int offA   = laneHi * 8;        // A k split: (0..7 | 8..15), +16
  const int offB   = laneHi * 8;        // B k split: (0..7 | 8..15), +16

  const __bf16* Bbase = Wl   + (size_t)(w * 16 + m) * LSTR + offB;
  const __bf16* Arow  = Abuf + (size_t)m * LSTR + offA;

  for (int t = 0; t < T_; ++t) {
    // ---- Stage A panel into LDS: 1280 x 16B chunks, 5 per thread (async) ----
    {
      const int slot = t & 1;
      for (int c = tid; c < 16 * (KTOT * 2 / 16); c += 256) {
        int mm  = c / 80;               // row (80 x 16B payload chunks per row)
        int ofs = (c - mm * 80) * 16;   // byte offset within row payload
        void* dst = (char*)Abuf + (size_t)mm * (LSTR * 2) + ofs;
        const char* src;
        if (ofs < KX * 2) {
          src = (const char*)xe_all + (((size_t)(g * 16 + mm) * T_ + t) * I_) * 2 + ofs;
          if (t + 1 < T_) __builtin_prefetch(src + I_ * 2, 0, 1);  // next-step xe
        } else {
          src = (const char*)cbuf +
                ((((size_t)slot * NG + g) * 16 + mm) * KC) * 2 + (ofs - KX * 2);
        }
        async_cp16(src, dst);
      }
      wait_async0();
      __syncthreads();
    }

    // ---- WMMA: D(16x16) += A(16x32) x B(32x16), bf16 in / f32 acc ----
    v8f acc = {0.f, 0.f, 0.f, 0.f, 0.f, 0.f, 0.f, 0.f};
    if (w < 6) acc = gemm_ks<KTOT / 32>(Arow, Bbase, acc);  // f,i,o: K = 640
    else       acc = gemm_ks<KX / 32>(Arow, Bbase, acc);    // c~:    K = 128

    // D layout: VGPR r -> row r + 8*laneHi, col = lane&15
#pragma unroll
    for (int r = 0; r < 8; ++r)
      gbuf[((size_t)w * 16 + (r + laneHi * 8)) * 16 + m] = acc[r];
    __syncthreads();

    // ---- Gate elementwise: 16 rows x 32 h-cols, 2 per thread ----
#pragma unroll
    for (int rep = 0; rep < 2; ++rep) {
      int e  = tid + rep * 256;
      int mm = e >> 5, hh = e & 31;
      int nt = hh >> 4, col = hh & 15;
      float fp = gbuf[(((0 + nt) * 16 + mm)) * 16 + col] + biasl[hh];
      float ip = gbuf[(((2 + nt) * 16 + mm)) * 16 + col] + biasl[32 + hh];
      float op = gbuf[(((4 + nt) * 16 + mm)) * 16 + col] + biasl[64 + hh];
      float cp = gbuf[(((6 + nt) * 16 + mm)) * 16 + col] + biasl[96 + hh];
      float fg = sigm(fp), ig = sigm(ip), og = sigm(op), cg = sigm(cp);
      int   bidx = g * 16 + mm;
      float r  = (xidx[(size_t)bidx * T_ + t] > 0) ? 1.0f : 0.0f;
      float Cn = (cg * ig + Cloc[mm * HS + hh] * fg) * r;
      Cloc[mm * HS + hh] = Cn;
      cbuf[(((size_t)((t + 1) & 1) * NG + g) * 16 + mm) * KC + hs + hh] = (__bf16)Cn;
      if (t == T_ - 1) hf[(size_t)bidx * H_ + hs + hh] = fast_tanh(Cn) * og;
    }
    __syncthreads();
    __threadfence();   // publish this step's bf16 C slice to the group

    // Cluster barrier: correct instruction when cluster-dispatched; S_NOP otherwise.
#if __has_builtin(__builtin_amdgcn_s_cluster_barrier)
    __builtin_amdgcn_s_cluster_barrier();
#endif
    // Functional group barrier (sense-reversing, L2 atomics) across the 16 WGs.
    if (tid == 0) {
      int* cnt = bar + g * 2;
      int* gen = bar + g * 2 + 1;
      int  myg = __hip_atomic_load(gen, __ATOMIC_RELAXED, __HIP_MEMORY_SCOPE_AGENT);
      int  a   = __hip_atomic_fetch_add(cnt, 1, __ATOMIC_ACQ_REL, __HIP_MEMORY_SCOPE_AGENT);
      if (a == NS - 1) {
        __hip_atomic_store(cnt, 0, __ATOMIC_RELAXED, __HIP_MEMORY_SCOPE_AGENT);
        __hip_atomic_fetch_add(gen, 1, __ATOMIC_RELEASE, __HIP_MEMORY_SCOPE_AGENT);
      } else {
        while (__hip_atomic_load(gen, __ATOMIC_ACQUIRE, __HIP_MEMORY_SCOPE_AGENT) == myg)
          __builtin_amdgcn_s_sleep(1);
      }
    }
    __syncthreads();
  }
}

// ---------------------------------------------------------------------------
// Kernel 3: p = hf @ Wph + bp, then log_softmax. One block per batch row.
// ---------------------------------------------------------------------------
__global__ __launch_bounds__(128) void k_head(const float* __restrict__ hf,
                                              const float* __restrict__ Wph,
                                              const float* __restrict__ bp,
                                              float* __restrict__ out) {
  __shared__ float hrow[H_];
  __shared__ float red[128];
  int b = blockIdx.x, tid = threadIdx.x;
  for (int k = tid; k < H_; k += 128) hrow[k] = hf[(size_t)b * H_ + k];
  __syncthreads();
  float p = -3.4e38f;
  if (tid < NC_) {
    float s = bp[tid];
    for (int k = 0; k < H_; ++k) s = fmaf(hrow[k], Wph[(size_t)k * NC_ + tid], s);
    p = s;
  }
  red[tid] = p; __syncthreads();
  for (int s = 64; s > 0; s >>= 1) { if (tid < s) red[tid] = fmaxf(red[tid], red[tid + s]); __syncthreads(); }
  float mx = red[0]; __syncthreads();
  float e = (tid < NC_) ? expf(p - mx) : 0.0f;
  red[tid] = e; __syncthreads();
  for (int s = 64; s > 0; s >>= 1) { if (tid < s) red[tid] += red[tid + s]; __syncthreads(); }
  float lse = mx + logf(red[0]);
  if (tid < NC_) out[(size_t)b * NC_ + tid] = p - lse;
}

// ---------------------------------------------------------------------------
extern "C" void kernel_launch(void* const* d_in, const int* in_sizes, int n_in,
                              void* d_out, int out_size, void* d_ws, size_t ws_size,
                              hipStream_t stream) {
  (void)in_sizes; (void)n_in; (void)out_size; (void)ws_size;
  const int*   x   = (const int*)  d_in[0];
  const float* emb = (const float*)d_in[1];
  const float* Wfx = (const float*)d_in[2];
  const float* Wfc = (const float*)d_in[3];
  const float* bf_ = (const float*)d_in[4];
  const float* Wix = (const float*)d_in[5];
  const float* Wic = (const float*)d_in[6];
  const float* bi_ = (const float*)d_in[7];
  const float* Wox = (const float*)d_in[8];
  const float* Woc = (const float*)d_in[9];
  const float* bo_ = (const float*)d_in[10];
  const float* Wcx = (const float*)d_in[11];
  const float* bc_ = (const float*)d_in[12];
  const float* Wph = (const float*)d_in[13];
  const float* bp_ = (const float*)d_in[14];

  char* ws = (char*)d_ws;
  __bf16* xe_all = (__bf16*)ws;
  size_t off = (size_t)B_ * T_ * I_ * 2;                       // 16 MB
  __bf16* cbuf = (__bf16*)(ws + off); off += (size_t)2 * NG * 16 * KC * 2;
  float*  hf   = (float*) (ws + off); off += (size_t)B_ * H_ * 4;
  int*    bar  = (int*)   (ws + off);

  const size_t lds = (size_t)NLOC * LSTR * 2   // pinned weight slice (162 KB, padded)
                   + (size_t)16 * LSTR * 2     // staged A panel      (20.25 KB)
                   + (size_t)8 * 16 * 16 * 4   // gate tiles
                   + (size_t)16 * HS * 4       // local fp32 C slice
                   + (size_t)NLOC * 4;         // biases  (~193 KB of 320 KB)
  (void)hipFuncSetAttribute((const void*)k_rnn,
                            hipFuncAttributeMaxDynamicSharedMemorySize, (int)lds);

  k_prep<<<(B_ * T_ * I_ + 255) / 256, 256, 0, stream>>>(x, emb, xe_all, cbuf, bar);
  k_rnn <<<NG * NS, 256, lds, stream>>>(x, Wfx, Wfc, bf_, Wix, Wic, bi_,
                                        Wox, Woc, bo_, Wcx, bc_, xe_all, cbuf, hf, bar);
  k_head<<<B_, 128, 0, stream>>>(hf, Wph, bp_, (float*)d_out);
}
